// MambaSpike_86990267613967
// MI455X (gfx1250) — matrix operand order repaired and checked by reading, not compiled, not run on hardware
//
#include <hip/hip_runtime.h>
#include <hip/hip_bf16.h>

// ---------------------------------------------------------------------------
// MambaSpike forward for MI455X (gfx1250, wave32, WMMA).
// GEMMs: v_wmma_f32_16x16x32_f16, one 32x32 macro-tile per wave (4 WMMAs per
// K-step). Operands are staged in f16: A row-major f16 (fragment = 2x b128),
// B pair-interleaved dwords matching the WMMA B VGPR layout (fragment = 8x
// coalesced b32, zero converts in the inner loop).
// Sequential parts (LIF frontend, selective scan) are scalar VALU kernels.
// ---------------------------------------------------------------------------

#define B_  8
#define T_  64
#define CIN 2
#define HID 32
#define SPKC 64
#define DM  512
#define DI  1024      // D_INNER
#define DS  16        // D_STATE
#define DTR 32        // DT_RANK
#define FEAT 16384    // 64 * 16 * 16
#define NCLS 11

typedef __attribute__((ext_vector_type(16))) _Float16 v16h;
typedef __attribute__((ext_vector_type(8)))  _Float16 v8h;
typedef __attribute__((ext_vector_type(8)))  float    v8f;

// ---------------------------------------------------------------------------
// Operand staging
// ---------------------------------------------------------------------------

// pack B (K x N, f32) -> pair-interleaved dwords: out[k/2][n] = {B[k][n], B[k+1][n]}
__global__ void pack_b_k(const float* __restrict__ in, unsigned* __restrict__ out,
                         long Kh, int N)
{
    long i = (long)blockIdx.x * blockDim.x + threadIdx.x;
    if (i >= Kh * N) return;
    long ko = i / N;
    int  n  = (int)(i - ko * N);
    union { _Float16 h[2]; unsigned u; } t;
    t.h[0] = (_Float16)in[(2 * ko) * (long)N + n];
    t.h[1] = (_Float16)in[(2 * ko + 1) * (long)N + n];
    out[i] = t.u;
}

// ---------------------------------------------------------------------------
// GEMM: D[M,N] = act( A[M,K] @ B[K,N] + bias ) + resid   (A f16, B packed f16)
// One wave -> one 32x32 macro-tile; M,N multiples of 32; K multiple of 32.
// act: 0 = none, 1 = softplus.  D16: optional f16 shadow of D (same ldd).
// WMMA fragment layouts used here (wave32, cf. CDNA5 ISA 7.12.2):
//   A (16x32): lane L: m=L&15; per-lane data = 2 contiguous 8-half runs.
//   B (32x16): lane L: n=L&15; VGPR v = pair K=hi*16+2v,+1 (code's k order, same as A) -> 1 dword.
//   D (16x16 f32): lane L, vgpr r: m = r + 8*hi, n = L&15.
// ---------------------------------------------------------------------------
__global__ void gemm_wmma_f16(const _Float16* __restrict__ A, int lda,
                              const unsigned* __restrict__ Bp, int ldbp, // = N
                              float* __restrict__ D, int ldd,
                              _Float16* __restrict__ D16,
                              int M, int N, int K,
                              const float* __restrict__ bias,
                              const float* __restrict__ resid,
                              int act)
{
    int wave   = blockIdx.x * (blockDim.x >> 5) + (threadIdx.x >> 5);
    int ntiles = N >> 5;
    int tiles  = (M >> 5) * ntiles;
    if (wave >= tiles) return;
    int mt = wave / ntiles;
    int nt = wave - mt * ntiles;

    int lane = threadIdx.x & 31;
    int mr   = lane & 15;
    int nc   = lane & 15;
    int hi   = lane >> 4;

    const _Float16* Arow0 = A + (long)(mt * 32 + mr) * lda;
    const _Float16* Arow1 = Arow0 + (long)16 * lda;
    const unsigned* Bp0   = Bp + nt * 32 + nc;
    const unsigned* Bp1   = Bp0 + 16;

    v8f acc00 = {}, acc01 = {}, acc10 = {}, acc11 = {};

    union AF { v16h v; v8h h[2]; };
    union BF { v16h v; unsigned u[8]; };

    for (int k0 = 0; k0 < K; k0 += 32) {
        if (k0 + 32 < K) {                       // prefetch next K slab of A
            __builtin_prefetch(Arow0 + k0 + 32, 0, 1);
            __builtin_prefetch(Arow1 + k0 + 32, 0, 1);
        }
        AF a0, a1; BF b0, b1;
        a0.h[0] = *(const v8h*)(Arow0 + k0 + hi * 16);
        a0.h[1] = *(const v8h*)(Arow0 + k0 + 8 + hi * 16);
        a1.h[0] = *(const v8h*)(Arow1 + k0 + hi * 16);
        a1.h[1] = *(const v8h*)(Arow1 + k0 + 8 + hi * 16);
        long kp = (long)((k0 >> 1) + hi * 8);
#pragma unroll
        for (int v = 0; v < 8; ++v) {
            b0.u[v] = Bp0[(kp + v) * ldbp];
            b1.u[v] = Bp1[(kp + v) * ldbp];
        }
        acc00 = __builtin_amdgcn_wmma_f32_16x16x32_f16(
            false, a0.v, false, b0.v, (short)0, acc00, false, false);
        acc01 = __builtin_amdgcn_wmma_f32_16x16x32_f16(
            false, a0.v, false, b1.v, (short)0, acc01, false, false);
        acc10 = __builtin_amdgcn_wmma_f32_16x16x32_f16(
            false, a1.v, false, b0.v, (short)0, acc10, false, false);
        acc11 = __builtin_amdgcn_wmma_f32_16x16x32_f16(
            false, a1.v, false, b1.v, (short)0, acc11, false, false);
    }

    int gm0 = mt * 32;
    int gn0 = nt * 32 + nc;
    v8f accs[4] = { acc00, acc01, acc10, acc11 };
#pragma unroll
    for (int q = 0; q < 4; ++q) {
        int gmB = gm0 + (q >> 1) * 16;
        int gn  = gn0 + (q & 1) * 16;
#pragma unroll
        for (int r = 0; r < 8; ++r) {
            int gm = gmB + r + 8 * hi;
            float val = accs[q][r];
            if (bias)     val += bias[gn];
            if (act == 1) val = (val > 20.f) ? val : log1pf(__expf(val));
            if (resid)    val += resid[(long)gm * ldd + gn];
            D[(long)gm * ldd + gn] = val;
            if (D16) D16[(long)gm * ldd + gn] = (_Float16)val;
        }
    }
}

// ---------------------------------------------------------------------------
// Frontend: conv3x3 (SAME) [+ maxpool2] + LIF, one kernel per stage per t.
// ---------------------------------------------------------------------------
__global__ void zero_k(float* __restrict__ p, long n)
{
    long i = (long)blockIdx.x * blockDim.x + threadIdx.x;
    if (i < n) p[i] = 0.f;
}

__device__ __forceinline__ float lif_update(float cur, float* mem, int idx)
{
    float m = mem[idx];
    float reset = (m > 1.0f) ? 1.0f : 0.0f;
    m = 0.9f * m + cur - reset;         // THR = 1.0
    mem[idx] = m;
    return (m - 1.0f > 0.0f) ? 1.0f : 0.0f;
}

// x:(B,T,2,64,64) -> conv 32ch -> pool -> LIF -> s1:(B,32,32,32)
__global__ void conv1_k(const float* __restrict__ x, const float* __restrict__ w,
                        const float* __restrict__ bias, float* __restrict__ mem,
                        float* __restrict__ s1, int t)
{
    int i  = blockIdx.x * blockDim.x + threadIdx.x;   // B*32*32*32
    int px = i & 31, py = (i >> 5) & 31, o = (i >> 10) & 31, b = i >> 15;
    const float* xin = x + (long)(b * T_ + t) * CIN * 4096;
    float mx = -1e30f;
    for (int sy = 0; sy < 2; ++sy)
    for (int sx = 0; sx < 2; ++sx) {
        int y = py * 2 + sy, xx = px * 2 + sx;
        float acc = bias[o];
        for (int ci = 0; ci < CIN; ++ci)
            for (int dy = 0; dy < 3; ++dy) {
                int yy = y + dy - 1; if (yy < 0 || yy >= 64) continue;
                for (int dx = 0; dx < 3; ++dx) {
                    int x2 = xx + dx - 1; if (x2 < 0 || x2 >= 64) continue;
                    acc += w[((o * CIN + ci) * 3 + dy) * 3 + dx]
                         * xin[ci * 4096 + yy * 64 + x2];
                }
            }
        mx = fmaxf(mx, acc);
    }
    s1[i] = lif_update(mx, mem, i);
}

// s1:(B,32,32,32) -> conv 32ch -> pool -> LIF -> s2:(B,32,16,16)
__global__ void conv2_k(const float* __restrict__ s1, const float* __restrict__ w,
                        const float* __restrict__ bias, float* __restrict__ mem,
                        float* __restrict__ s2)
{
    int i  = blockIdx.x * blockDim.x + threadIdx.x;   // B*32*16*16
    int px = i & 15, py = (i >> 4) & 15, o = (i >> 8) & 31, b = i >> 13;
    const float* xin = s1 + (long)b * HID * 1024;
    float mx = -1e30f;
    for (int sy = 0; sy < 2; ++sy)
    for (int sx = 0; sx < 2; ++sx) {
        int y = py * 2 + sy, xx = px * 2 + sx;
        float acc = bias[o];
        for (int ci = 0; ci < HID; ++ci)
            for (int dy = 0; dy < 3; ++dy) {
                int yy = y + dy - 1; if (yy < 0 || yy >= 32) continue;
                for (int dx = 0; dx < 3; ++dx) {
                    int x2 = xx + dx - 1; if (x2 < 0 || x2 >= 32) continue;
                    acc += w[((o * HID + ci) * 3 + dy) * 3 + dx]
                         * xin[ci * 1024 + yy * 32 + x2];
                }
            }
        mx = fmaxf(mx, acc);
    }
    s2[i] = lif_update(mx, mem, i);
}

// s2:(B,32,16,16) -> conv 64ch -> LIF -> spk[b,t,c*256+y*16+x]
__global__ void conv3_k(const float* __restrict__ s2, const float* __restrict__ w,
                        const float* __restrict__ bias, float* __restrict__ mem,
                        float* __restrict__ spk, int t)
{
    int i  = blockIdx.x * blockDim.x + threadIdx.x;   // B*64*16*16
    int px = i & 15, py = (i >> 4) & 15, o = (i >> 8) & 63, b = i >> 14;
    const float* xin = s2 + (long)b * HID * 256;
    float acc = bias[o];
    for (int ci = 0; ci < HID; ++ci)
        for (int dy = 0; dy < 3; ++dy) {
            int yy = py + dy - 1; if (yy < 0 || yy >= 16) continue;
            for (int dx = 0; dx < 3; ++dx) {
                int x2 = px + dx - 1; if (x2 < 0 || x2 >= 16) continue;
                acc += w[((o * HID + ci) * 3 + dy) * 3 + dx]
                     * xin[ci * 256 + yy * 16 + x2];
            }
        }
    float s = lif_update(acc, mem, i);
    spk[((long)(b * T_ + t) * SPKC + o) * 256 + py * 16 + px] = s;
}

// act16[b,t,f] = mean of spk[b, t-2 .. t+2, f] (zero padded) / 5, stored f16
__global__ void rate_smooth_k(const float* __restrict__ spk,
                              _Float16* __restrict__ act16)
{
    long i = (long)blockIdx.x * blockDim.x + threadIdx.x;  // B*T*FEAT
    if (i >= (long)B_ * T_ * FEAT) return;
    int  f  = (int)(i & (FEAT - 1));
    long bt = i >> 14;                 // FEAT = 2^14
    int  t  = (int)(bt & (T_ - 1));
    long b  = bt >> 6;
    float s = 0.f;
#pragma unroll
    for (int k = -2; k <= 2; ++k) {
        int tt = t + k;
        if (tt >= 0 && tt < T_) s += spk[((b << 6) + tt) * (long)FEAT + f];
    }
    act16[i] = (_Float16)(s * 0.2f);
}

// ---------------------------------------------------------------------------
// Mamba helpers
// ---------------------------------------------------------------------------
// layernorm, f32 in -> f16 out (consumed only as GEMM A operand)
__global__ void layernorm_rows(const float* __restrict__ X, _Float16* __restrict__ Y,
                               const float* __restrict__ w, const float* __restrict__ b,
                               int ncols)
{
    int row = blockIdx.x;
    const float* x = X + (long)row * ncols;
    _Float16* y = Y + (long)row * ncols;
    __shared__ float red[256];
    float s = 0.f;
    for (int c = threadIdx.x; c < ncols; c += blockDim.x) s += x[c];
    red[threadIdx.x] = s; __syncthreads();
    for (int st = 128; st > 0; st >>= 1) {
        if ((int)threadIdx.x < st) red[threadIdx.x] += red[threadIdx.x + st];
        __syncthreads();
    }
    float mu = red[0] / ncols; __syncthreads();
    float v = 0.f;
    for (int c = threadIdx.x; c < ncols; c += blockDim.x) {
        float d = x[c] - mu; v += d * d;
    }
    red[threadIdx.x] = v; __syncthreads();
    for (int st = 128; st > 0; st >>= 1) {
        if ((int)threadIdx.x < st) red[threadIdx.x] += red[threadIdx.x + st];
        __syncthreads();
    }
    float rstd = rsqrtf(red[0] / ncols + 1e-5f);
    for (int c = threadIdx.x; c < ncols; c += blockDim.x)
        y[c] = (_Float16)((x[c] - mu) * rstd * w[c] + b[c]);
}

// depthwise causal conv (width 4) over time + SiLU; xi = xr[:, :DI]
// writes f32 (scan input) and f16 shadow (GEMM A operand)
__global__ void depthconv_silu_k(const float* __restrict__ xr,
                                 const float* __restrict__ w,   // (DI,4)
                                 float* __restrict__ xc,
                                 _Float16* __restrict__ xc16)
{
    long i = (long)blockIdx.x * blockDim.x + threadIdx.x;   // B*T*DI
    if (i >= (long)B_ * T_ * DI) return;
    int  d   = (int)(i & (DI - 1));
    long row = i >> 10;
    int  t   = (int)(row & (T_ - 1));
    float acc = 0.f;
#pragma unroll
    for (int k = 0; k < 4; ++k) {
        int tt = t + k - 3;
        if (tt >= 0)
            acc += w[d * 4 + k] * xr[(row + (tt - t)) * (long)(2 * DI) + d];
    }
    acc = acc / (1.f + __expf(-acc));   // SiLU
    xc[row * (long)DI + d]   = acc;
    xc16[row * (long)DI + d] = (_Float16)acc;
}

// selective scan: one lane per (b,d); Bm/Cm staged in LDS per block (same b).
// grid = (DI/256, B), block = 256; output f16 (GEMM A operand only)
__global__ void mamba_scan_k(const float* __restrict__ dt,    // (B*T, DI)
                             const float* __restrict__ xdbl,  // (B*T, 64): [32:48)=B, [48:64)=C
                             const float* __restrict__ xc,    // (B*T, DI)
                             const float* __restrict__ xr,    // (B*T, 2*DI), res at +DI
                             const float* __restrict__ A_log, // (DI, DS)
                             const float* __restrict__ Dp,    // (DI)
                             _Float16* __restrict__ Y16)      // (B*T, DI)
{
    int d = blockIdx.x * blockDim.x + threadIdx.x;
    int b = blockIdx.y;
    float A[DS], hs[DS];
#pragma unroll
    for (int n = 0; n < DS; ++n) {
        A[n]  = -__expf(A_log[d * DS + n]);
        hs[n] = 0.f;
    }
    float Dd = Dp[d];
    __shared__ float sB[DS], sC[DS];
    for (int t = 0; t < T_; ++t) {
        long row = (long)(b * T_ + t);
        if (threadIdx.x < DS)
            sB[threadIdx.x] = xdbl[row * 64 + 32 + threadIdx.x];
        else if (threadIdx.x < 2 * DS)
            sC[threadIdx.x - DS] = xdbl[row * 64 + 32 + threadIdx.x];
        __syncthreads();
        float dtv = dt[row * DI + d];
        float u   = xc[row * DI + d];
        float yv  = 0.f;
#pragma unroll
        for (int n = 0; n < DS; ++n) {
            float dA = __expf(dtv * A[n]);
            hs[n] = dA * hs[n] + dtv * sB[n] * u;
            yv += hs[n] * sC[n];
        }
        float res  = xr[row * (long)(2 * DI) + DI + d];
        float sres = res / (1.f + __expf(-res));
        Y16[row * DI + d] = (_Float16)((yv + u * Dd) * sres);
        __syncthreads();
    }
}

// mean over T, layernorm, classifier. grid=B, block=512
__global__ void finalize_k(const float* __restrict__ h,
                           const float* __restrict__ fnw, const float* __restrict__ fnb,
                           const float* __restrict__ clsw, const float* __restrict__ clsb,
                           float* __restrict__ out)
{
    int b = blockIdx.x, tid = threadIdx.x;
    __shared__ float s[DM], red[DM];
    float m = 0.f;
    for (int t = 0; t < T_; ++t) m += h[(long)(b * T_ + t) * DM + tid];
    m *= (1.f / (float)T_);
    s[tid] = m;
    red[tid] = m; __syncthreads();
    for (int st = 256; st > 0; st >>= 1) {
        if (tid < st) red[tid] += red[tid + st];
        __syncthreads();
    }
    float mu = red[0] / (float)DM; __syncthreads();
    float dv = s[tid] - mu;
    red[tid] = dv * dv; __syncthreads();
    for (int st = 256; st > 0; st >>= 1) {
        if (tid < st) red[tid] += red[tid + st];
        __syncthreads();
    }
    float rstd = rsqrtf(red[0] / (float)DM + 1e-5f); __syncthreads();
    s[tid] = (s[tid] - mu) * rstd * fnw[tid] + fnb[tid];
    __syncthreads();
    if (tid < NCLS) {
        float acc = clsb[tid];
        for (int k = 0; k < DM; ++k) acc += s[k] * clsw[k * NCLS + tid];
        out[b * NCLS + tid] = acc;
    }
}

// ---------------------------------------------------------------------------
// Launch
// ---------------------------------------------------------------------------
static inline void gemm_launch(const _Float16* A, int lda,
                               const unsigned* Bp, int N_,
                               float* D, int ldd, _Float16* D16,
                               int M, int N, int K,
                               const float* bias, const float* resid, int act,
                               hipStream_t stream)
{
    int tiles  = (M / 32) * (N / 32);
    int blocks = (tiles + 7) / 8;     // 8 waves (32x32 macro-tiles) per block
    gemm_wmma_f16<<<blocks, 256, 0, stream>>>(A, lda, Bp, N_, D, ldd, D16,
                                              M, N, K, bias, resid, act);
}

static inline void pack_launch(const float* src, unsigned* dst, long Ktot, int N,
                               hipStream_t stream)
{
    long n = (Ktot / 2) * (long)N;
    int blocks = (int)((n + 255) / 256);
    pack_b_k<<<blocks, 256, 0, stream>>>(src, dst, Ktot / 2, N);
}

extern "C" void kernel_launch(void* const* d_in, const int* in_sizes, int n_in,
                              void* d_out, int out_size, void* d_ws, size_t ws_size,
                              hipStream_t stream)
{
    const float* x      = (const float*)d_in[0];
    const float* c1w    = (const float*)d_in[1];
    const float* c1b    = (const float*)d_in[2];
    const float* c2w    = (const float*)d_in[3];
    const float* c2b    = (const float*)d_in[4];
    const float* c3w    = (const float*)d_in[5];
    const float* c3b    = (const float*)d_in[6];
    const float* inp_w  = (const float*)d_in[7];
    const float* inp_b  = (const float*)d_in[8];
    const float* ln_w   = (const float*)d_in[9];
    const float* ln_b   = (const float*)d_in[10];
    const float* in_w   = (const float*)d_in[11];
    const float* conv_w = (const float*)d_in[12];
    const float* xprj_w = (const float*)d_in[13];
    const float* dt_w   = (const float*)d_in[14];
    const float* dt_b   = (const float*)d_in[15];
    const float* A_log  = (const float*)d_in[16];
    const float* Dp     = (const float*)d_in[17];
    const float* out_w  = (const float*)d_in[18];
    const float* fn_w   = (const float*)d_in[19];
    const float* fn_b   = (const float*)d_in[20];
    const float* cls_w  = (const float*)d_in[21];
    const float* cls_b  = (const float*)d_in[22];

    char* base = (char*)d_ws;
    size_t off = 0;
    auto carveB = [&](size_t bytes) {
        void* p = base + off;
        off += (bytes + 255) & ~(size_t)255;
        return p;
    };
    auto carveF = [&](size_t n) { return (float*)carveB(n * 4); };
    auto carveH = [&](size_t n) { return (_Float16*)carveB(n * 2); };
    auto carveU = [&](size_t n) { return (unsigned*)carveB(n * 4); };

    // f32 buffers
    float* mem1 = carveF((size_t)B_ * HID * 32 * 32);     // 262144
    float* mem2 = carveF((size_t)B_ * HID * 16 * 16);     //  65536
    float* mem3 = carveF((size_t)B_ * SPKC * 16 * 16);    // 131072
    float* s1   = carveF((size_t)B_ * HID * 32 * 32);
    float* s2   = carveF((size_t)B_ * HID * 16 * 16);
    float* spk  = carveF((size_t)B_ * T_ * FEAT);
    float* h    = carveF((size_t)B_ * T_ * DM);
    float* h2   = carveF((size_t)B_ * T_ * DM);
    float* xr   = carveF((size_t)B_ * T_ * 2 * DI);
    float* xc   = carveF((size_t)B_ * T_ * DI);
    float* xdbl = carveF((size_t)B_ * T_ * 64);
    float* dtb  = carveF((size_t)B_ * T_ * DI);
    // f16 activation shadows (GEMM A operands)
    _Float16* act16  = carveH((size_t)B_ * T_ * FEAT);
    _Float16* hln16  = carveH((size_t)B_ * T_ * DM);
    _Float16* xc16   = carveH((size_t)B_ * T_ * DI);
    _Float16* xdbl16 = carveH((size_t)B_ * T_ * 64);
    _Float16* yb16   = carveH((size_t)B_ * T_ * DI);
    // packed f16 weights (GEMM B operands), pair-interleaved dwords
    unsigned* pw_inp = carveU((size_t)(FEAT / 2) * DM);          // 4.19M
    unsigned* pw_in  = carveU((size_t)4 * (DM / 2) * 2 * DI);    // 2.10M
    unsigned* pw_xp  = carveU((size_t)4 * (DI / 2) * 64);
    unsigned* pw_dt  = carveU((size_t)4 * (DTR / 2) * DI);
    unsigned* pw_out = carveU((size_t)4 * (DI / 2) * DM);

    // ---- stage weights to f16 packed (K even per layer -> safe to flatten) --
    pack_launch(inp_w,  pw_inp, FEAT,     DM,     stream);
    pack_launch(in_w,   pw_in,  4 * DM,   2 * DI, stream);
    pack_launch(xprj_w, pw_xp,  4 * DI,   64,     stream);
    pack_launch(dt_w,   pw_dt,  4 * DTR,  DI,     stream);
    pack_launch(out_w,  pw_out, 4 * DI,   DM,     stream);

    // zero LIF membranes (mem1..mem3 are contiguous): 458752 floats
    zero_k<<<1792, 256, 0, stream>>>(mem1, 458752L);

    // ---- spiking frontend: 64 sequential timesteps ----
    for (int t = 0; t < T_; ++t) {
        conv1_k<<<1024, 256, 0, stream>>>(x, c1w, c1b, mem1, s1, t);
        conv2_k<<< 256, 256, 0, stream>>>(s1, c2w, c2b, mem2, s2);
        conv3_k<<< 512, 256, 0, stream>>>(s2, c3w, c3b, mem3, spk, t);
    }

    // ---- rate smoothing (writes f16 GEMM operand) ----
    rate_smooth_k<<<32768, 256, 0, stream>>>(spk, act16);

    // ---- input projection: h = act @ inp_w + inp_b  (512 x 16384 x 512) ----
    gemm_launch(act16, FEAT, pw_inp, DM, h, DM, nullptr,
                B_ * T_, DM, FEAT, inp_b, nullptr, 0, stream);

    // ---- 4 Mamba blocks ----
    for (int l = 0; l < 4; ++l) {
        const float* lnw = ln_w  + (size_t)l * DM;
        const float* lnb = ln_b  + (size_t)l * DM;
        const float* cw  = conv_w + (size_t)l * DI * 4;
        const float* db  = dt_b  + (size_t)l * DI;
        const float* Al  = A_log + (size_t)l * DI * DS;
        const float* Dl  = Dp    + (size_t)l * DI;
        const unsigned* piw = pw_in  + (size_t)l * (DM / 2) * 2 * DI;
        const unsigned* pxw = pw_xp  + (size_t)l * (DI / 2) * 64;
        const unsigned* pdw = pw_dt  + (size_t)l * (DTR / 2) * DI;
        const unsigned* pow_ = pw_out + (size_t)l * (DI / 2) * DM;

        layernorm_rows<<<B_ * T_, 256, 0, stream>>>(h, hln16, lnw, lnb, DM);

        // xr = hln @ in_w  (512 x 512 x 2048)
        gemm_launch(hln16, DM, piw, 2 * DI, xr, 2 * DI, nullptr,
                    B_ * T_, 2 * DI, DM, nullptr, nullptr, 0, stream);

        // depthwise causal conv + SiLU (f32 + f16 shadow)
        depthconv_silu_k<<<2048, 256, 0, stream>>>(xr, cw, xc, xc16);

        // x_dbl = xc @ xproj_w  (512 x 1024 x 64), f16 shadow for dt GEMM
        gemm_launch(xc16, DI, pxw, 64, xdbl, 64, xdbl16,
                    B_ * T_, 64, DI, nullptr, nullptr, 0, stream);

        // dt = softplus(x_dbl[:, :32] @ dt_w + dt_b)  (512 x 32 x 1024)
        gemm_launch(xdbl16, 64, pdw, DI, dtb, DI, nullptr,
                    B_ * T_, DI, DTR, db, nullptr, 1, stream);

        // selective scan + gating (writes f16 GEMM operand)
        mamba_scan_k<<<dim3(DI / 256, B_), 256, 0, stream>>>(
            dtb, xdbl, xc, xr, Al, Dl, yb16);

        // h2 = h + y @ out_w  (512 x 1024 x 512, residual)
        gemm_launch(yb16, DI, pow_, DM, h2, DM, nullptr,
                    B_ * T_, DM, DI, nullptr, h, 0, stream);

        float* tmp = h; h = h2; h2 = tmp;
    }

    // ---- mean over T, layernorm, classifier ----
    finalize_k<<<B_, 512, 0, stream>>>(h, fn_w, fn_b, cls_w, cls_b,
                                       (float*)d_out);
}
